// ActionNetSo3_80461917323980
// MI455X (gfx1250) — compile-verified, matrix operand order checked by reading, not run on hardware
//
#include <hip/hip_runtime.h>

// ---------------------------------------------------------------------------
// MI455X (gfx1250) Wigner-D block-diagonal apply.
// Store-bound: 256 MiB out @ 23.3 TB/s ~= 11.5 us floor; all compute (~5 GFLOP)
// is negligible -> f16 WMMA (probe-confirmed builtin), D built in f32,
// coefficient table in f64 once. item_rep staged to LDS via the Tensor Data
// Mover (TDM) in WMMA-fragment order; D written straight into fragment order.
// ---------------------------------------------------------------------------

#define DEGREES 15
#define MDIM    256
#define REP     64
#define NBATCH  4096

typedef __attribute__((ext_vector_type(16))) _Float16     v16h;
typedef __attribute__((ext_vector_type(8)))  float        v8f;
typedef __attribute__((ext_vector_type(4)))  unsigned int v4u;
typedef __attribute__((ext_vector_type(8)))  int          v8i;
typedef __attribute__((ext_vector_type(4)))  int          v4i;

// prefix of (2j+1)^3 (A-table block offsets, floats)
__device__ __constant__ int c_off3[16] =
  {0,1,28,153,496,1225,2556,4753,8128,13041,19900,29161,41328,56953,76636,101025};
// prefix of (2j+1)^2 (dim^2 pair offsets)
__device__ __constant__ int c_pref2[17] =
  {0,1,10,35,84,165,286,455,680,969,1330,1771,2300,2925,3654,4495,5456};

// ---------------------------------------------------------------------------
// Kernel 1: Wigner small-d coefficient table A[l][mp][m][p] (f32, 523264 B).
// One thread per (l,mp,m); f64 factorials.
// ---------------------------------------------------------------------------
__global__ void prep_wigner_A(float* __restrict__ wsA) {
  int idx = blockIdx.x * blockDim.x + threadIdx.x;
  if (idx >= 5456) return;
  int l = 0;
  while (c_pref2[l + 1] <= idx) ++l;
  const int dim = 2 * l + 1;
  int local = idx - c_pref2[l];
  int mp_i = local / dim, m_i = local % dim;
  int mp = mp_i - l, m = m_i - l;

  double fact[32];
  fact[0] = 1.0;
  for (int i = 1; i < 32; ++i) fact[i] = fact[i - 1] * (double)i;
  double pref = sqrt(fact[l + mp] * fact[l - mp] * fact[l + m] * fact[l - m]);

  double acc[31];
  for (int p = 0; p < dim; ++p) acc[p] = 0.0;
  int kmin = (m - mp > 0) ? (m - mp) : 0;
  int kmax = (l + m < l - mp) ? (l + m) : (l - mp);
  for (int k = kmin; k <= kmax; ++k) {
    double denom = fact[l + m - k] * fact[k] * fact[l - mp - k] * fact[mp - m + k];
    double sgn = ((mp - m + k) & 1) ? -1.0 : 1.0;
    int p = 2 * l + m - mp - 2 * k;
    acc[p] += sgn * pref / denom;
  }
  float* dst = wsA + c_off3[l] + (size_t)(mp_i * dim + m_i) * dim;
  for (int p = 0; p < dim; ++p) dst[p] = (float)acc[p];
}

// ---------------------------------------------------------------------------
// WMMA 16-bit fragment K-indexing (documented A layout; B assumed mirrored):
//   element e (0..15) of lane (hi = lane>=16):
//   v = e>>1;  K = ((v<4) ? 2v : 2v+8) + 8*hi + (e&1)
// ---------------------------------------------------------------------------
__device__ __forceinline__ int frag_k(int e, int hi) {
  int v = e >> 1;
  return ((v < 4) ? (v << 1) : ((v << 1) + 8)) + (hi << 3) + (e & 1);
}

// ---------------------------------------------------------------------------
// Kernel 2: item_rep -> f16 B *fragment-ordered* layout (64 KB):
//   wsB[(((l*4 + ct)*32 + lane)*16 + e)] = B(k, n) of K-padded (32-row) block l
// so each lane's 16 halves are contiguous (one 32-B LDS/global vector load).
// ---------------------------------------------------------------------------
__global__ void prep_rep_b16(const float* __restrict__ rep, _Float16* __restrict__ wsB) {
  int idx = blockIdx.x * blockDim.x + threadIdx.x;   // 32768 total
  int e    = idx & 15;
  int lane = (idx >> 4) & 31;
  int ct   = (idx >> 9) & 3;
  int l    = idx >> 11;
  int hi = lane >> 4, lp = lane & 15;
  int k = frag_k(e, hi);                 // padded row within block (0..31)
  int n = ct * 16 + lp;                  // rep column
  int dim = 2 * l + 1, st = l * l;
  float v = (k < dim) ? rep[(st + k) * REP + n] : 0.0f;
  wsB[idx] = (_Float16)v;
}

// ---------------------------------------------------------------------------
// Kernel 3: per batch element, build real Wigner-D blocks straight into
// fragment-ordered LDS (f16) and WMMA them against the TDM-staged B.
// 256 threads = 8 waves; 24 (l,rowtile) pairs * 4 coltiles = 96 WMMA tiles.
// ---------------------------------------------------------------------------
__launch_bounds__(256)
__global__ void wigner_apply(const float* __restrict__ angles,
                             const float* __restrict__ wsA,
                             const _Float16* __restrict__ wsB,
                             float* __restrict__ out) {
  // A fragments: 24 pairs x 32 lanes x 16 halves (24 KB)
  __shared__ __align__(32) _Float16 sAf[24 * 32 * 16];
  // B fragments staged by the TDM: 16 l x 4 ct x 32 lanes x 16 halves (64 KB)
  __shared__ __align__(32) _Float16 sB[32768];
  __shared__ float sd[31 * 31];   // small-d block of current l
  __shared__ float spt[32];       // cos^p sin^(2l-p) table

  const int tid = threadIdx.x;
  const int n = blockIdx.x;

  // ---- wave 0: kick off the async TDM copy of wsB (64 KB) into LDS.
  // It overlaps the entire D-construction below; EXEC is ignored by TDM,
  // the branch keeps it to a single wave (one descriptor, one DMA).
  if (tid < 32) {
    unsigned lds_off = (unsigned)(uintptr_t)(void*)sB;     // low 32 bits = LDS offset
    unsigned long long ga = (unsigned long long)(uintptr_t)wsB;
    v4u g0 = { 1u,                                  // count=1 (valid descriptor)
               lds_off,                             // lds_addr
               (unsigned)ga,                        // global_addr[31:0]
               (unsigned)(ga >> 32) | (2u << 30) }; // global_addr[56:32] | type=2
    v8i g1 = { (int)(1u << 16),        // workgroup_mask=0, data_size=1 (2 bytes)
               (int)0x80000000u,       // tensor_dim0 = 32768 (lo16 in [31:16])
               (int)(1u << 16),        // tensor_dim0 hi=0 | tensor_dim1 = 1
               (int)0x80000000u,       // tensor_dim1 hi=0 | tile_dim0 = 32768
               0,                      // tile_dim1 = 0, tile_dim2 = 0 (1-D tile)
               32768,                  // tensor_dim0_stride lo32
               0, 0 };
    v4i g2 = { 0, 0, 0, 0 };
    v4i g3 = { 0, 0, 0, 0 };
#if __has_include(<hip/amd_detail/amd_gfx1250_TDM.h>)
    v8i gx = { 0, 0, 0, 0, 0, 0, 0, 0 };
    __builtin_amdgcn_tensor_load_to_lds(g0, g1, g2, g3, gx, 0);
#else
    __builtin_amdgcn_tensor_load_to_lds(g0, g1, g2, g3, 0);
#endif
  }

  const float alpha = angles[3 * n + 0];
  const float beta  = angles[3 * n + 1];
  const float gamma = angles[3 * n + 2];
  const float cb = cosf(0.5f * beta), sb = sinf(0.5f * beta);

  // zero the A-fragment region (covers all padding)
  {
    int* sAi = (int*)sAf;
    for (int i = tid; i < 6144; i += 256) sAi[i] = 0;
  }
  __syncthreads();

  const float isq = 0.70710678118654752f;

  for (int l = 0; l <= DEGREES; ++l) {
    const int dim = 2 * l + 1;

    // power table pt[p] = cb^p * sb^(2l-p)
    if (tid < dim) spt[tid] = powf(cb, (float)tid) * powf(sb, (float)(2 * l - tid));
    __syncthreads();

    // small-d: d[mp][m] = sum_p A[mp][m][p] * pt[p]
    const float* Al = wsA + c_off3[l];
    for (int e = tid; e < dim * dim; e += 256) {
      const float* ap = Al + (size_t)e * dim;
      float sum = 0.0f;
      for (int p = 0; p < dim; ++p) sum += ap[p] * spt[p];
      sd[e] = sum;
    }
    __syncthreads();

    // real-basis D, written straight into A-fragment order:
    //   rt = a>>4, lane = (a&15) + 16*hi with hi=(K - matches layout),
    //   K = b;  K' = K - 8*hi;  e = (K'>=16) ? K'-8 : K'
    for (int ee = tid; ee < dim * dim; ee += 256) {
      int a = ee / dim, b = ee % dim;
      int mu = a - l, nu = b - l;

      float mm[2], ur[2], ui[2];
      if (mu > 0) {
        mm[0] = (float)(-mu); ur[0] = isq;  ui[0] = 0.f;
        mm[1] = (float)( mu); ur[1] = (mu & 1) ? -isq : isq; ui[1] = 0.f;
      } else if (mu < 0) {
        int u = -mu;
        mm[0] = (float)(-u);  ur[0] = 0.f;  ui[0] = isq;
        mm[1] = (float)( u);  ur[1] = 0.f;  ui[1] = (u & 1) ? isq : -isq;
      } else {
        mm[0] = 0.f; ur[0] = 1.f; ui[0] = 0.f;
        mm[1] = 0.f; ur[1] = 0.f; ui[1] = 0.f;
      }
      float kkm[2], vr[2], vi[2];
      if (nu > 0) {
        kkm[0] = (float)(-nu); vr[0] = isq;  vi[0] = 0.f;
        kkm[1] = (float)( nu); vr[1] = (nu & 1) ? -isq : isq; vi[1] = 0.f;
      } else if (nu < 0) {
        int v = -nu;
        kkm[0] = (float)(-v);  vr[0] = 0.f;  vi[0] = -isq;
        kkm[1] = (float)( v);  vr[1] = 0.f;  vi[1] = (v & 1) ? -isq : isq;
      } else {
        kkm[0] = 0.f; vr[0] = 1.f; vi[0] = 0.f;
        kkm[1] = 0.f; vr[1] = 0.f; vi[1] = 0.f;
      }

      float acc = 0.0f;
#pragma unroll
      for (int ii = 0; ii < 2; ++ii)
#pragma unroll
        for (int jj = 0; jj < 2; ++jj) {
          float cr = ur[ii] * vr[jj] - ui[ii] * vi[jj];
          float ci = ur[ii] * vi[jj] + ui[ii] * vr[jj];
          float th = alpha * mm[ii] + gamma * kkm[jj];
          int mi = (int)mm[ii] + l, ki = (int)kkm[jj] + l;
          acc += sd[mi * dim + ki] * (cr * cosf(th) + ci * sinf(th));
        }

      // (a,b) -> fragment slot
      int rt = a >> 4;
      int hiK = (b >> 3) & 1;
      int kp = b - (hiK << 3);
      int e = (kp >= 16) ? (kp - 8) : kp;
      int lane = (a & 15) + (hiK << 4);
      int pair = (l <= 7) ? l : (8 + ((l - 8) << 1) + rt);
      sAf[((pair * 32 + lane) << 4) + e] = (_Float16)acc;
    }
    __syncthreads();
  }

  // ---- wave 0 makes the TDM data visible, barrier publishes to all waves.
  if (tid < 32) __builtin_amdgcn_s_wait_tensorcnt(0);
  __syncthreads();

  // ---- WMMA stage: wave-uniform control flow (EXEC all 1s at each WMMA) ----
  const int w = tid >> 5, lane = tid & 31;
  const int hi = lane >> 4, lp = lane & 15;

  for (int j = 0; j < 3; ++j) {
    int p = w + 8 * j;                 // pair 0..23
    int l, rt;
    if (p < 8) { l = p; rt = 0; }
    else       { int q = p - 8; l = 8 + (q >> 1); rt = q & 1; }
    const int dim = 2 * l + 1, st = l * l;

    // A fragment: one contiguous 32-byte LDS vector load
    v16h af = *(const v16h*)(sAf + ((p * 32 + lane) << 4));

#pragma unroll 1
    for (int ct = 0; ct < 4; ++ct) {
      // B fragment: one contiguous 32-byte LDS vector load
      v16h bf = *(const v16h*)(sB + ((((l << 2) + ct) * 32 + lane) << 4));

      v8f cz = {0.f, 0.f, 0.f, 0.f, 0.f, 0.f, 0.f, 0.f};
      v8f d = __builtin_amdgcn_wmma_f32_16x16x32_f16(
          /*neg_a=*/false, af, /*neg_b=*/false, bf,
          /*c_mod=*/(short)0, cz, /*reuse_a=*/false, /*reuse_b=*/false);
#pragma unroll
      for (int i = 0; i < 8; ++i) {
        int rb = rt * 16 + i + hi * 8;   // row within block
        if (rb < dim) {
          out[(size_t)n * (MDIM * REP) + (size_t)(st + rb) * REP + ct * 16 + lp] = d[i];
        }
      }
    }
  }
}

// ---------------------------------------------------------------------------
extern "C" void kernel_launch(void* const* d_in, const int* in_sizes, int n_in,
                              void* d_out, int out_size, void* d_ws, size_t ws_size,
                              hipStream_t stream) {
  const float* angles = (const float*)d_in[0];    // (4096,3) f32
  const float* rep    = (const float*)d_in[1];    // (256,64) f32
  float* out = (float*)d_out;                     // (4096, 16384) f32

  float*    wsA = (float*)d_ws;                            // 523264 B
  _Float16* wsB = (_Float16*)((char*)d_ws + 523264);       // 65536 B, fragment order

  prep_wigner_A<<<(5456 + 255) / 256, 256, 0, stream>>>(wsA);
  prep_rep_b16<<<128, 256, 0, stream>>>(rep, wsB);
  wigner_apply<<<NBATCH, 256, 0, stream>>>(angles, wsA, wsB, out);
}
